// GinModel_77309411650
// MI455X (gfx1250) — matrix operand order repaired and checked by reading, not compiled, not run on hardware
//
#include <hip/hip_runtime.h>

typedef __attribute__((ext_vector_type(16))) _Float16 v16h;
typedef __attribute__((ext_vector_type(8)))  float    v8f;

#define NN        100000
#define EE        1600000
#define NGRAPH    16
#define PER_G     (EE / NGRAPH)    // 100000 edges per graph segment
#define NODE_THR  (NN / NGRAPH)    // 6250 threshold step
#define NTILES    (NN / 16)        // 6250 16-node tiles
#define TPB_TILES 8
#define MLP_BLOCKS ((NTILES + TPB_TILES - 1) / TPB_TILES)

__device__ __forceinline__ void gAdd(float* p, float v) {
  (void)__hip_atomic_fetch_add(p, v, __ATOMIC_RELAXED, __HIP_MEMORY_SCOPE_AGENT);
}
__device__ __forceinline__ void sAdd(float* p, float v) {
  (void)__hip_atomic_fetch_add(p, v, __ATOMIC_RELAXED, __HIP_MEMORY_SCOPE_WORKGROUP);
}

// ---------------------------------------------------------------- zero
__global__ void zero_kernel(float* p, int n) {
  int i = blockIdx.x * blockDim.x + threadIdx.x;
  int stride = gridDim.x * blockDim.x;
  for (; i < n; i += stride) p[i] = 0.f;
}

// ------------------------------------------------- pack weights into WMMA B layout
// B (KxNn, row-major f32) -> f16 operands of 32x16, per-lane contiguous:
// element e: op = e/512, lane = (e/16)%32, j = e%16
// k = kc*32 + (lane>>4)*8 + (j&7) + (j>>3)*16 ; n = nc*16 + (lane&15) ; op = nc*KC + kc
__global__ void pack_kernel(const float* __restrict__ W, int K, int Nn, _Float16* __restrict__ dst) {
  int total = K * Nn;
  int KC = K / 32;
  for (int e = blockIdx.x * blockDim.x + threadIdx.x; e < total; e += gridDim.x * blockDim.x) {
    int op   = e >> 9;
    int lane = (e >> 4) & 31;
    int j    = e & 15;
    int kc = op % KC;
    int nc = op / KC;
    int k = kc * 32 + ((lane >> 4) << 3) + (j & 7) + ((j >> 3) << 4);
    int n = nc * 16 + (lane & 15);
    dst[e] = (_Float16)W[k * Nn + n];
  }
}

// ---------------------------------------------------------------- edge scatter-add
// one wave per edge; lane handles 2 features. Validity is wave-uniform.
template <bool BNIN>
__global__ __launch_bounds__(256) void edge_agg_kernel(const float* __restrict__ feat,
                                                       const int* __restrict__ ei,
                                                       const float* __restrict__ sc,
                                                       const float* __restrict__ sh,
                                                       float* __restrict__ agg) {
  const int lane = threadIdx.x & 31;
  const int e = blockIdx.x * 8 + (threadIdx.x >> 5);
  const int src = ei[e];
  const int dst = ei[EE + e];
  const int thr = (e / PER_G) * NODE_THR;
  if (src >= thr && dst >= thr) {
    const int f = lane * 2;
    float2 v = *(const float2*)(feat + (size_t)src * 64 + f);
    if constexpr (BNIN) {
      float2 s = *(const float2*)(sc + f);
      float2 t = *(const float2*)(sh + f);
      v.x = fmaf(v.x, s.x, t.x);
      v.y = fmaf(v.y, s.y, t.y);
    }
    float* d = agg + (size_t)dst * 64 + f;
    gAdd(d, v.x);
    gAdd(d + 1, v.y);
  }
}

// ---------------------------------------------------------------- WMMA helpers
template <bool BNIN>
__device__ __forceinline__ void loadGrp(const float* rx, const float* ra,
                                        const float* sc, const float* sh,
                                        int k, float4& h) {
  float4 xv = *(const float4*)(rx + k);
  float4 av = *(const float4*)(ra + k);
  if constexpr (BNIN) {
    float4 s = *(const float4*)(sc + k);
    float4 t = *(const float4*)(sh + k);
    h.x = fmaf(xv.x, s.x, t.x) + av.x;
    h.y = fmaf(xv.y, s.y, t.y) + av.y;
    h.z = fmaf(xv.z, s.z, t.z) + av.z;
    h.w = fmaf(xv.w, s.w, t.w) + av.w;
  } else {
    h.x = xv.x + av.x; h.y = xv.y + av.y; h.z = xv.z + av.z; h.w = xv.w + av.w;
  }
}

__device__ __forceinline__ v16h packA(const float4& h0, const float4& h1,
                                      const float4& h2, const float4& h3) {
  v16h A;
  A[0]  = (_Float16)h0.x; A[1]  = (_Float16)h0.y; A[2]  = (_Float16)h0.z; A[3]  = (_Float16)h0.w;
  A[4]  = (_Float16)h1.x; A[5]  = (_Float16)h1.y; A[6]  = (_Float16)h1.z; A[7]  = (_Float16)h1.w;
  A[8]  = (_Float16)h2.x; A[9]  = (_Float16)h2.y; A[10] = (_Float16)h2.z; A[11] = (_Float16)h2.w;
  A[12] = (_Float16)h3.x; A[13] = (_Float16)h3.y; A[14] = (_Float16)h3.z; A[15] = (_Float16)h3.w;
  return A;
}

// A operand (16x32 f16): lane row m=l&15, K runs {kb..kb+7, kb+16..kb+23}, kb=kc*32+hk
template <bool BNIN>
__device__ __forceinline__ v16h buildA_global(const float* rx, const float* ra,
                                              const float* sc, const float* sh, int kb) {
  float4 h0, h1, h2, h3;
  loadGrp<BNIN>(rx, ra, sc, sh, kb,      h0);
  loadGrp<BNIN>(rx, ra, sc, sh, kb + 4,  h1);
  loadGrp<BNIN>(rx, ra, sc, sh, kb + 16, h2);
  loadGrp<BNIN>(rx, ra, sc, sh, kb + 20, h3);
  return packA(h0, h1, h2, h3);
}

__device__ __forceinline__ v16h buildA_lds(const float* row, int kb) {
  float4 h0 = *(const float4*)(row + kb);
  float4 h1 = *(const float4*)(row + kb + 4);
  float4 h2 = *(const float4*)(row + kb + 16);
  float4 h3 = *(const float4*)(row + kb + 20);
  return packA(h0, h1, h2, h3);
}

// ---------------------------------------------------------------- fused MLP (2 GEMMs + ReLU + BN stats)
// one wave computes a 16-node tile: z = relu( relu(H @ W1 + b1) @ W2 + b2 )
template <int DMID, bool BNIN>
__global__ __launch_bounds__(256) void mlp_kernel(const float* xin,
                                                  const float* __restrict__ agg,
                                                  const float* __restrict__ sIn,
                                                  const float* __restrict__ tIn,
                                                  const _Float16* __restrict__ pW1,
                                                  const float* __restrict__ b1,
                                                  const _Float16* __restrict__ pW2,
                                                  const float* __restrict__ b2,
                                                  float* zout,
                                                  float* __restrict__ bnSum,
                                                  float* __restrict__ bnSq) {
  constexpr int NCM = DMID / 16;  // mid N tiles
  constexpr int KC2 = DMID / 32;  // gemm2 K tiles
  constexpr int STR = DMID + 4;   // LDS row stride (bank-conflict-free, 16B aligned)

  __shared__ float lbuf[TPB_TILES * 16 * STR];
  __shared__ float bnS[64];
  __shared__ float bnQ[64];

  const int tid = threadIdx.x;
  if (tid < 64) { bnS[tid] = 0.f; bnQ[tid] = 0.f; }
  __syncthreads();

  const int lane = tid & 31;
  const int wv   = tid >> 5;
  const int tile = blockIdx.x * TPB_TILES + wv;
  const int m    = lane & 15;
  const int hi   = lane >> 4;
  const int hk   = hi << 3;

  if (tile < NTILES) {  // wave-uniform: EXEC stays all-ones for WMMA
    const int node = tile * 16 + m;
    const float* rx = xin + (size_t)node * 64;
    const float* ra = agg + (size_t)node * 64;

    // ---- GEMM1: (16x64) @ (64xDMID) ----
    v8f acc[NCM];
    #pragma unroll
    for (int nc = 0; nc < NCM; ++nc) {
      float bv = b1[nc * 16 + m];
      #pragma unroll
      for (int r = 0; r < 8; ++r) acc[nc][r] = bv;
    }
    #pragma unroll
    for (int kc = 0; kc < 2; ++kc) {
      v16h A = buildA_global<BNIN>(rx, ra, sIn, tIn, kc * 32 + hk);
      #pragma unroll
      for (int nc = 0; nc < NCM; ++nc) {
        v16h B = *(const v16h*)(pW1 + (size_t)(nc * 2 + kc) * 512 + lane * 16);
        acc[nc] = __builtin_amdgcn_wmma_f32_16x16x32_f16(false, A, false, B,
                                                         (short)0, acc[nc], false, false);
      }
    }

    // ---- ReLU + per-wave LDS transpose (C layout -> A layout) ----
    float* my = lbuf + wv * 16 * STR;
    #pragma unroll
    for (int nc = 0; nc < NCM; ++nc) {
      #pragma unroll
      for (int r = 0; r < 8; ++r) {
        my[(r + 8 * hi) * STR + nc * 16 + m] = fmaxf(acc[nc][r], 0.f);
      }
    }

    // ---- GEMM2: (16xDMID) @ (DMIDx64) ----
    v8f acc2[4];
    #pragma unroll
    for (int nc = 0; nc < 4; ++nc) {
      float bv = b2[nc * 16 + m];
      #pragma unroll
      for (int r = 0; r < 8; ++r) acc2[nc][r] = bv;
    }
    const float* mrow = my + m * STR;
    #pragma unroll
    for (int kc = 0; kc < KC2; ++kc) {
      v16h A2 = buildA_lds(mrow, kc * 32 + hk);
      #pragma unroll
      for (int nc = 0; nc < 4; ++nc) {
        v16h B = *(const v16h*)(pW2 + (size_t)(nc * KC2 + kc) * 512 + lane * 16);
        acc2[nc] = __builtin_amdgcn_wmma_f32_16x16x32_f16(false, A2, false, B,
                                                          (short)0, acc2[nc], false, false);
      }
    }

    // ---- ReLU + store z + BN partial stats ----
    float* zr = zout + (size_t)tile * 16 * 64;
    #pragma unroll
    for (int nc = 0; nc < 4; ++nc) {
      float s = 0.f, q = 0.f;
      #pragma unroll
      for (int r = 0; r < 8; ++r) {
        float v = fmaxf(acc2[nc][r], 0.f);
        zr[(r + 8 * hi) * 64 + nc * 16 + m] = v;
        s += v;
        q += v * v;
      }
      s += __shfl_xor(s, 16, 32);
      q += __shfl_xor(q, 16, 32);
      if (hi == 0) {
        sAdd(&bnS[nc * 16 + m], s);
        sAdd(&bnQ[nc * 16 + m], q);
      }
    }
  }

  __syncthreads();
  if (tid < 64) {
    gAdd(bnSum + tid, bnS[tid]);
    gAdd(bnSq + tid, bnQ[tid]);
  }
}

// ---------------------------------------------------------------- BN finalize: fold into scale/shift
__global__ void bn_finalize_kernel(const float* __restrict__ sum, const float* __restrict__ sq,
                                   const float* __restrict__ gamma, const float* __restrict__ beta,
                                   float* __restrict__ sOut, float* __restrict__ tOut) {
  int f = threadIdx.x;
  if (f < 64) {
    float mean = sum[f] * (1.0f / NN);
    float var  = fmaxf(sq[f] * (1.0f / NN) - mean * mean, 0.f);
    float sc   = gamma[f] * rsqrtf(var + 1e-5f);
    sOut[f] = sc;
    tOut[f] = beta[f] - mean * sc;
  }
}

// ---------------------------------------------------------------- mean pool (batch is sorted)
__global__ __launch_bounds__(256) void pool_kernel(const float* __restrict__ z,
                                                   const int* __restrict__ batch,
                                                   const float* __restrict__ sc,
                                                   const float* __restrict__ sh,
                                                   float* __restrict__ pool,
                                                   float* __restrict__ cnt) {
  __shared__ float lp[NGRAPH * 64];
  __shared__ float lc[NGRAPH];
  int tid = threadIdx.x;
  for (int i = tid; i < NGRAPH * 64; i += 256) lp[i] = 0.f;
  if (tid < NGRAPH) lc[tid] = 0.f;
  __syncthreads();

  int base = blockIdx.x * 64;
  int node = base + (tid >> 2);
  int fs = (tid & 3) * 16;
  if (node < NN) {
    int g = batch[node];
    const float* zr = z + (size_t)node * 64 + fs;
    #pragma unroll
    for (int u = 0; u < 16; u += 4) {
      float4 v = *(const float4*)(zr + u);
      float4 s = *(const float4*)(sc + fs + u);
      float4 t = *(const float4*)(sh + fs + u);
      sAdd(&lp[g * 64 + fs + u + 0], fmaf(v.x, s.x, t.x));
      sAdd(&lp[g * 64 + fs + u + 1], fmaf(v.y, s.y, t.y));
      sAdd(&lp[g * 64 + fs + u + 2], fmaf(v.z, s.z, t.z));
      sAdd(&lp[g * 64 + fs + u + 3], fmaf(v.w, s.w, t.w));
    }
    if ((tid & 3) == 0) sAdd(&lc[g], 1.0f);
  }
  __syncthreads();

  int last = base + 63;
  if (last > NN - 1) last = NN - 1;
  int gmin = batch[base];
  int gmax = batch[last];
  for (int g = gmin; g <= gmax; ++g) {
    if (tid < 64) {
      float v = lp[g * 64 + tid];
      if (v != 0.f) gAdd(pool + (size_t)g * 64 + tid, v);
    } else if (tid == 64) {
      float c = lc[g];
      if (c != 0.f) gAdd(cnt + g, c);
    }
  }
}

__global__ void out_kernel(const float* __restrict__ pool, const float* __restrict__ cnt,
                           float* __restrict__ out) {
  int i = blockIdx.x * 256 + threadIdx.x;
  if (i < NGRAPH * 64) out[i] = pool[i] / fmaxf(cnt[i >> 6], 1.0f);
}

// ---------------------------------------------------------------- launch
extern "C" void kernel_launch(void* const* d_in, const int* in_sizes, int n_in,
                              void* d_out, int out_size, void* d_ws, size_t ws_size,
                              hipStream_t stream) {
  const float* x   = (const float*)d_in[0];
  const int* ei    = (const int*)d_in[1];
  const int* batch = (const int*)d_in[2];
  const float* W1a = (const float*)d_in[5];
  const float* b1a = (const float*)d_in[6];
  const float* W1b = (const float*)d_in[7];
  const float* b1b = (const float*)d_in[8];
  const float* g1  = (const float*)d_in[9];
  const float* be1 = (const float*)d_in[10];
  const float* W2a = (const float*)d_in[11];
  const float* b2a = (const float*)d_in[12];
  const float* W2b = (const float*)d_in[13];
  const float* b2b = (const float*)d_in[14];
  const float* g2  = (const float*)d_in[15];
  const float* be2 = (const float*)d_in[16];

  char* ws = (char*)d_ws;
  float* bufA = (float*)ws;                              // N*64 aggregation buffer
  float* bufB = (float*)(ws + (size_t)NN * 64 * 4);      // N*64 activation buffer
  char* sm = ws + (size_t)2 * NN * 64 * 4;
  _Float16* pk = (_Float16*)sm;                          // 12288 packed f16 weights
  float* fsm = (float*)(sm + 24576);
  float* bn1S = fsm +   0; float* bn1Q = fsm +  64;
  float* bn2S = fsm + 128; float* bn2Q = fsm + 192;
  float* s1   = fsm + 256; float* t1   = fsm + 320;
  float* s2   = fsm + 384; float* t2   = fsm + 448;
  float* pool = fsm + 512;                               // 16*64
  float* cnt  = fsm + 1536;                              // 16
  _Float16* pW1a = pk;
  _Float16* pW1b = pk + 2048;
  _Float16* pW2a = pk + 4096;
  _Float16* pW2b = pk + 8192;

  // pack weights into WMMA-B lane layout (f16)
  pack_kernel<<<8, 256, 0, stream>>>(W1a, 64, 32, pW1a);
  pack_kernel<<<8, 256, 0, stream>>>(W1b, 32, 64, pW1b);
  pack_kernel<<<16, 256, 0, stream>>>(W2a, 64, 64, pW2a);
  pack_kernel<<<16, 256, 0, stream>>>(W2b, 64, 64, pW2b);

  // zero aggregation + stats/pool accumulators
  zero_kernel<<<2048, 256, 0, stream>>>(bufA, NN * 64);
  zero_kernel<<<8, 256, 0, stream>>>(fsm, 1552);

  // ---- layer 1 ----
  edge_agg_kernel<false><<<EE / 8, 256, 0, stream>>>(x, ei, nullptr, nullptr, bufA);
  mlp_kernel<32, false><<<MLP_BLOCKS, 256, 0, stream>>>(x, bufA, nullptr, nullptr,
                                                        pW1a, b1a, pW1b, b1b,
                                                        bufB, bn1S, bn1Q);
  bn_finalize_kernel<<<1, 64, 0, stream>>>(bn1S, bn1Q, g1, be1, s1, t1);

  // ---- layer 2 (BN folded into scale/shift on the fly) ----
  zero_kernel<<<2048, 256, 0, stream>>>(bufA, NN * 64);
  edge_agg_kernel<true><<<EE / 8, 256, 0, stream>>>(bufB, ei, s1, t1, bufA);
  mlp_kernel<64, true><<<MLP_BLOCKS, 256, 0, stream>>>(bufB, bufA, s1, t1,
                                                       pW2a, b2a, pW2b, b2b,
                                                       bufB, bn2S, bn2Q);
  bn_finalize_kernel<<<1, 64, 0, stream>>>(bn2S, bn2Q, g2, be2, s2, t2);

  // ---- pool + output ----
  pool_kernel<<<(NN + 63) / 64, 256, 0, stream>>>(bufB, batch, s2, t2, pool, cnt);
  out_kernel<<<4, 256, 0, stream>>>(pool, cnt, (float*)d_out);
}
